// LSTM_42880953483455
// MI455X (gfx1250) — matrix operand order repaired and checked by reading, not compiled, not run on hardware
//
#include <hip/hip_runtime.h>
#include <hip/hip_bf16.h>
#include <math.h>

typedef __bf16 bf16;
typedef __attribute__((ext_vector_type(16))) __bf16 v16bf;
typedef __attribute__((ext_vector_type(8)))  __bf16 v8bf;
typedef __attribute__((ext_vector_type(8)))  float  v8f;

#define B_  64
#define T_  512
#define D_  512
#define U_  512
#define G_  2048          // 4*U_
#define K_  1024          // D_ (x part) + U_ (h part)
#define KB_ (K_ / 32)     // 32 k-blocks
#define UB  16            // hidden units per workgroup
#define NWG 32            // U_/UB
#define NTH 256           // 8 waves (wave32)

#define KPAD   1032       // bf16 elems per weight column incl pad (2064B, 16B aligned)
#define APITCH 40         // bf16 elems per A-panel row incl pad (80B)

#define W_BYTES  (2 * 64 * KPAD * 2)       // hi/lo weight slabs in LDS
#define A_PREC_B (64 * APITCH * 2)         // one precision slab of A staging
#define A_BUF_B  (2 * A_PREC_B)            // one double-buffer entry (hi+lo)
#define A_BYTES  (2 * A_BUF_B)             // double buffered
#define G_BYTES  (4 * B_ * UB * 4)         // gate exchange (fp32)
#define SMEM_BYTES (W_BYTES + A_BYTES + G_BYTES)

__device__ __forceinline__ float hsig(float v) {
  return fminf(fmaxf(0.2f * v + 0.5f, 0.0f), 1.0f);
}

// CDNA5 async global->LDS copy (VGLOBAL, tracked by ASYNCcnt).
// LDS destination is the LDS base plus the vdst VGPR value; generic->LDS addr = low 32 bits (aperture trunc).
__device__ __forceinline__ void async_g2l_b128(unsigned lds_off, const void* gptr) {
  asm volatile("global_load_async_to_lds_b128 %0, %1, off"
               :
               : "v"(lds_off), "v"(gptr)
               : "memory");
}

__device__ __forceinline__ void wait_asynccnt0() {
#if __has_builtin(__builtin_amdgcn_s_wait_asynccnt)
  __builtin_amdgcn_s_wait_asynccnt(0);
#else
  asm volatile("s_wait_asynccnt 0x0" ::: "memory");
#endif
}

#define WMMA_BF16(A, Bm, C) \
  __builtin_amdgcn_wmma_f32_16x16x32_bf16(false, (A), false, (Bm), (short)0, (C), false, false)

// ---------------------------------------------------------------------------
// Prep: bf16 hi/lo splits of x, W, U; zero h ping-pong buffers and barrier.
// ---------------------------------------------------------------------------
__global__ void lstm_prep(const float* __restrict__ x, const float* __restrict__ W,
                          const float* __restrict__ U,
                          bf16* __restrict__ x_hi, bf16* __restrict__ x_lo,
                          bf16* __restrict__ W_hi, bf16* __restrict__ W_lo,
                          bf16* __restrict__ U_hi, bf16* __restrict__ U_lo,
                          bf16* __restrict__ h_hi, bf16* __restrict__ h_lo,
                          unsigned* __restrict__ bar) {
  const long stride = (long)gridDim.x * blockDim.x;
  const long i0 = (long)blockIdx.x * blockDim.x + threadIdx.x;
  for (long i = i0; i < (long)B_ * T_ * D_; i += stride) {
    float v = x[i];
    bf16 hv = (bf16)v;
    x_hi[i] = hv;
    x_lo[i] = (bf16)(v - (float)hv);
  }
  for (long i = i0; i < (long)D_ * G_; i += stride) {
    float v = W[i];
    bf16 hv = (bf16)v;
    W_hi[i] = hv;
    W_lo[i] = (bf16)(v - (float)hv);
    v = U[i];
    hv = (bf16)v;
    U_hi[i] = hv;
    U_lo[i] = (bf16)(v - (float)hv);
  }
  for (long i = i0; i < 2L * B_ * U_; i += stride) {
    h_hi[i] = (bf16)0.0f;
    h_lo[i] = (bf16)0.0f;
  }
  if (i0 == 0) *bar = 0u;
}

// ---------------------------------------------------------------------------
// Persistent LSTM: 32 workgroups, weights stationary in LDS, grid-sync/step,
// async double-buffered A staging.
// ---------------------------------------------------------------------------
__launch_bounds__(NTH, 1)
__global__ void lstm_persistent(const bf16* __restrict__ x_hi, const bf16* __restrict__ x_lo,
                                const bf16* __restrict__ W_hi, const bf16* __restrict__ W_lo,
                                const bf16* __restrict__ U_hi, const bf16* __restrict__ U_lo,
                                const float* __restrict__ bias,
                                bf16* __restrict__ h_hi, bf16* __restrict__ h_lo,
                                unsigned* __restrict__ bar,
                                float* __restrict__ out) {
  extern __shared__ char smem[];
  bf16*  wlds = (bf16*)smem;                         // [prec][64 cols][KPAD] col-major, K contiguous
  bf16*  alds = (bf16*)(smem + W_BYTES);             // [buf][prec][64 rows][APITCH]
  float* gbuf = (float*)(smem + W_BYTES + A_BYTES);  // [4 gates][64 b][16 u]

  const int tid = threadIdx.x;
  const int wg  = blockIdx.x;
  const int u0  = wg * UB;

  // ---- one-time: load this WGP's weight slice into LDS (transposed) ----
  // combined B-matrix row k: k < 512 -> U[k][g] (recurrent), else W[k-512][g] (input)
  for (int prec = 0; prec < 2; ++prec) {
    const bf16* Um = prec ? U_lo : U_hi;
    const bf16* Wm = prec ? W_lo : W_hi;
    bf16* dst = wlds + prec * 64 * KPAD;
    for (int idx = tid; idx < 64 * K_; idx += NTH) {
      int c = idx & 63;
      int k = idx >> 6;
      int g = (c >> 4) * U_ + u0 + (c & 15);
      dst[c * KPAD + k] = (k < D_) ? Um[k * G_ + g] : Wm[(k - D_) * G_ + g];
    }
  }

  // per-thread bias + register-resident cell state (thread owns same (b,u) forever)
  float bi[4], bff[4], bcc[4], boo[4], creg[4];
  #pragma unroll
  for (int j = 0; j < 4; ++j) {
    int idx = tid + NTH * j;
    int u = u0 + (idx & 15);
    bi[j]  = bias[u];
    bff[j] = bias[U_ + u];
    bcc[j] = bias[2 * U_ + u];
    boo[j] = bias[3 * U_ + u];
    creg[j] = 0.0f;
  }

  const int wave   = tid >> 5;
  const int lane   = tid & 31;
  const int lane15 = lane & 15;
  const int gate   = wave & 3;         // which gate's N-tile this wave computes
  const int mhalf  = wave >> 2;        // which pair of M-tiles
  const int bcol   = gate * 16 + lane15;
  const int abase  = (lane >= 16) ? 8 : 0;    // A-operand K sub-offset (ISA 7.12.2)
  const int bbase  = (lane >= 16) ? 8 : 0;    // B-operand K sub-offset
  const int srow   = tid >> 2;         // A staging: row 0..63
  const int sq     = tid & 3;          // A staging: 16B quarter

  // row bases for this wave's two M-tiles
  const int m0 = (mhalf * 2 + 0) * 16 + lane15;
  const int m1 = m0 + 16;

  // LDS byte offsets for async destinations (wave-relative LDS address)
  const unsigned a_lds0 = (unsigned)(uintptr_t)alds;
  const unsigned st_off = (unsigned)(srow * APITCH + sq * 8) * 2u;

  __syncthreads();

  for (int t = 0; t < T_; ++t) {
    const int rd = t & 1;
    const bf16* hh = h_hi + rd * (B_ * U_);
    const bf16* hl = h_lo + rd * (B_ * U_);

    v8f acc0 = {};
    v8f acc1 = {};

    // async stage of A panel k-block kb into double buffer `buf`
    auto stage = [&](int kb, int buf) {
      const int k0 = kb * 32;
      const bf16 *gh, *gl;
      if (k0 < D_) {                               // recurrent part: h
        int o = srow * U_ + k0 + sq * 8;
        gh = hh + o;
        gl = hl + o;
      } else {                                     // input part: x[b][t][:]
        int o = (srow * T_ + t) * D_ + (k0 - D_) + sq * 8;
        gh = x_hi + o;
        gl = x_lo + o;
      }
      unsigned d = a_lds0 + (unsigned)buf * A_BUF_B + st_off;
      async_g2l_b128(d, gh);
      async_g2l_b128(d + A_PREC_B, gl);
    };

    // consume k-block kb from buffer `cur`: 12 ds_load_b128 + 6 WMMA
    auto consume = [&](int kb, int cur) {
      const int k0 = kb * 32;
      const bf16* ab = alds + cur * (A_BUF_B / 2);   // elem offset = bytes/2

      const bf16* bph = wlds + bcol * KPAD + k0 + bbase;
      const bf16* bpl = wlds + 64 * KPAD + bcol * KPAD + k0 + bbase;
      v8bf b0 = *(const v8bf*)bph;
      v8bf b1 = *(const v8bf*)(bph + 16);
      v8bf b2 = *(const v8bf*)bpl;
      v8bf b3 = *(const v8bf*)(bpl + 16);
      v16bf Bh = __builtin_shufflevector(b0, b1, 0,1,2,3,4,5,6,7,8,9,10,11,12,13,14,15);
      v16bf Bl = __builtin_shufflevector(b2, b3, 0,1,2,3,4,5,6,7,8,9,10,11,12,13,14,15);

      const bf16* a0p = ab + m0 * APITCH + abase;
      const bf16* a1p = ab + m1 * APITCH + abase;
      const bf16* a0q = ab + 64 * APITCH + m0 * APITCH + abase;
      const bf16* a1q = ab + 64 * APITCH + m1 * APITCH + abase;
      v8bf x0 = *(const v8bf*)a0p;
      v8bf x1 = *(const v8bf*)(a0p + 16);
      v8bf x2 = *(const v8bf*)a1p;
      v8bf x3 = *(const v8bf*)(a1p + 16);
      v8bf x4 = *(const v8bf*)a0q;
      v8bf x5 = *(const v8bf*)(a0q + 16);
      v8bf x6 = *(const v8bf*)a1q;
      v8bf x7 = *(const v8bf*)(a1q + 16);
      v16bf Ah0 = __builtin_shufflevector(x0, x1, 0,1,2,3,4,5,6,7,8,9,10,11,12,13,14,15);
      v16bf Ah1 = __builtin_shufflevector(x2, x3, 0,1,2,3,4,5,6,7,8,9,10,11,12,13,14,15);
      v16bf Al0 = __builtin_shufflevector(x4, x5, 0,1,2,3,4,5,6,7,8,9,10,11,12,13,14,15);
      v16bf Al1 = __builtin_shufflevector(x6, x7, 0,1,2,3,4,5,6,7,8,9,10,11,12,13,14,15);

      // bf16x3: hi*hi + hi*lo + lo*hi, acc0/acc1 interleaved (independent chains)
      acc0 = WMMA_BF16(Ah0, Bh, acc0);
      acc1 = WMMA_BF16(Ah1, Bh, acc1);
      acc0 = WMMA_BF16(Ah0, Bl, acc0);
      acc1 = WMMA_BF16(Ah1, Bl, acc1);
      acc0 = WMMA_BF16(Al0, Bh, acc0);
      acc1 = WMMA_BF16(Al1, Bh, acc1);
    };

    stage(0, 0);
    wait_asynccnt0();
    __syncthreads();

    #pragma unroll
    for (int kb = 0; kb < KB_ - 1; ++kb) {
      stage(kb + 1, (kb & 1) ^ 1);
      consume(kb, kb & 1);
      wait_asynccnt0();     // our stage for kb+1 has landed
      __syncthreads();      // everyone's stage for kb+1 has landed
    }
    consume(KB_ - 1, (KB_ - 1) & 1);

    // ---- exchange gate tiles through LDS (D-matrix layout, ISA 7.12.2) ----
    {
      int mt0 = (mhalf * 2 + 0) * 16 + ((lane < 16) ? 0 : 8);
      int mt1 = mt0 + 16;
      #pragma unroll
      for (int r = 0; r < 8; ++r) {
        gbuf[(gate * B_ + (mt0 + r)) * UB + lane15] = acc0[r];
        gbuf[(gate * B_ + (mt1 + r)) * UB + lane15] = acc1[r];
      }
    }
    __syncthreads();

    // ---- elementwise LSTM cell update; c stays in registers ----
    const int wr = (t + 1) & 1;
    bf16* nhh = h_hi + wr * (B_ * U_);
    bf16* nhl = h_lo + wr * (B_ * U_);
    #pragma unroll
    for (int j = 0; j < 4; ++j) {
      int idx  = tid + NTH * j;
      int brow = idx >> 4;
      int ul   = idx & 15;
      float gi = gbuf[(0 * B_ + brow) * UB + ul] + bi[j];
      float gf = gbuf[(1 * B_ + brow) * UB + ul] + bff[j];
      float gc = gbuf[(2 * B_ + brow) * UB + ul] + bcc[j];
      float go = gbuf[(3 * B_ + brow) * UB + ul] + boo[j];
      float c  = hsig(gf) * creg[j] + hsig(gi) * tanhf(gc);
      creg[j]  = c;
      float hn = hsig(go) * tanhf(c);
      int u = u0 + ul;
      out[(brow * T_ + t) * U_ + u] = hn;
      bf16 hhi = (bf16)hn;
      nhh[brow * U_ + u] = hhi;
      nhl[brow * U_ + u] = (bf16)(hn - (float)hhi);
    }

    // ---- device-wide step barrier (monotonic L2 atomic counter) ----
    __syncthreads();
    if (tid == 0) {
      __threadfence();
      __hip_atomic_fetch_add(bar, 1u, __ATOMIC_ACQ_REL, __HIP_MEMORY_SCOPE_AGENT);
      const unsigned target = (unsigned)(NWG * (t + 1));
      while (__hip_atomic_load(bar, __ATOMIC_ACQUIRE, __HIP_MEMORY_SCOPE_AGENT) < target) {
        __builtin_amdgcn_s_sleep(2);
      }
    }
    __syncthreads();
    __threadfence();
  }
}

// ---------------------------------------------------------------------------
extern "C" void kernel_launch(void* const* d_in, const int* in_sizes, int n_in,
                              void* d_out, int out_size, void* d_ws, size_t ws_size,
                              hipStream_t stream) {
  const float* x  = (const float*)d_in[0];   // [B,T,D]
  const float* W  = (const float*)d_in[1];   // [D,4U]
  const float* U  = (const float*)d_in[2];   // [U,4U]
  const float* bv = (const float*)d_in[3];   // [4U]
  float* out = (float*)d_out;                // [B,T,U]

  char* ws = (char*)d_ws;
  size_t off = 0;
  auto carve = [&](size_t bytes) -> char* {
    char* p = ws + off;
    off += (bytes + 255) & ~(size_t)255;
    return p;
  };
  const size_t nx = (size_t)B_ * T_ * D_;
  const size_t nw = (size_t)D_ * G_;
  bf16* x_hi = (bf16*)carve(nx * 2);
  bf16* x_lo = (bf16*)carve(nx * 2);
  bf16* W_hi = (bf16*)carve(nw * 2);
  bf16* W_lo = (bf16*)carve(nw * 2);
  bf16* U_hi = (bf16*)carve(nw * 2);
  bf16* U_lo = (bf16*)carve(nw * 2);
  bf16* h_hi = (bf16*)carve((size_t)2 * B_ * U_ * 2);
  bf16* h_lo = (bf16*)carve((size_t)2 * B_ * U_ * 2);
  unsigned* bar = (unsigned*)carve(256);

  lstm_prep<<<2048, NTH, 0, stream>>>(x, W, U, x_hi, x_lo, W_hi, W_lo,
                                      U_hi, U_lo, h_hi, h_lo, bar);
  lstm_persistent<<<NWG, NTH, SMEM_BYTES, stream>>>(x_hi, x_lo, W_hi, W_lo,
                                                    U_hi, U_lo, bv,
                                                    h_hi, h_lo, bar, out);
}
